// LSTMTrafficPredictor_36730560315502
// MI455X (gfx1250) — matrix operand order repaired and checked by reading, not compiled, not run on hardware
//
#include <hip/hip_runtime.h>
#include <hip/hip_bf16.h>

// ---- problem sizes ----
#define B_SZ 2048
#define T_SZ 512
#define H1 64
#define H2 32
#define G1 256   // 4*H1
#define G2 128   // 4*H2
#define FCN 16

typedef __attribute__((ext_vector_type(16))) __bf16 v16bf;
typedef __attribute__((ext_vector_type(8)))  __bf16 v8bf;
typedef __attribute__((ext_vector_type(8)))  float  v8f;
typedef __attribute__((ext_vector_type(2)))  float  v2f;

__device__ __forceinline__ float fast_tanh(float x) {
#if __has_builtin(__builtin_amdgcn_tanhf)
  return __builtin_amdgcn_tanhf(x);          // v_tanh_f32 (CDNA5 TRANS op)
#else
  return 2.0f * __builtin_amdgcn_rcpf(1.0f + __expf(-2.0f * x)) - 1.0f;
#endif
}
__device__ __forceinline__ float fast_sigmoid(float x) {
#if __has_builtin(__builtin_amdgcn_tanhf)
  return fmaf(0.5f, __builtin_amdgcn_tanhf(0.5f * x), 0.5f);
#else
  return __builtin_amdgcn_rcpf(1.0f + __expf(-x));
#endif
}

// 16x32 bf16 A-operand tile from a row-major [16][K] LDS buffer.
// rowbase = this lane's row (m = lane&15); koff = kbase + 16*(lane>>4).
__device__ __forceinline__ v16bf load_a_tile(const __bf16* rowbase, int koff) {
  v8bf c0 = *(const v8bf*)(rowbase + koff);
  v8bf c1 = *(const v8bf*)(rowbase + koff + 8);
  return __builtin_shufflevector(c0, c1,
      0,1,2,3,4,5,6,7,8,9,10,11,12,13,14,15);
}

__global__ void __launch_bounds__(128)
lstm2_fused_wmma(const float* __restrict__ x,
                 const float* __restrict__ Wih1, const float* __restrict__ Whh1,
                 const float* __restrict__ bih1, const float* __restrict__ bhh1,
                 const float* __restrict__ Wih2, const float* __restrict__ Whh2,
                 const float* __restrict__ bih2, const float* __restrict__ bhh2,
                 const float* __restrict__ fc1w, const float* __restrict__ fc1b,
                 const float* __restrict__ fc2w, const float* __restrict__ fc2b,
                 float* __restrict__ out)
{
  // 64 KB total: weights bf16 in B-operand-friendly [N][K] layout + ping-pong h-state
  __shared__ alignas(32) __bf16 w1h[G1 * H1];       // 32 KB  Whh1
  __shared__ alignas(32) __bf16 w2x[G2 * H1];       // 16 KB  Wih2
  __shared__ alignas(32) __bf16 w2h[G2 * H2];       //  8 KB  Whh2
  __shared__ alignas(32) __bf16 w1xs[G1 * 4];       //  2 KB  Wih1 (swizzled {0,2,1,3})
  __shared__ alignas(32) __bf16 hb1[2][16 * H1];    //  4 KB  h1 state ping-pong
  __shared__ alignas(32) __bf16 hb2[2][16 * H2];    //  2 KB  h2 state ping-pong

  const int tid   = threadIdx.x;
  const int w     = tid >> 5;          // wave id 0..3
  const int lane  = tid & 31;
  const int hi    = lane >> 4;
  const int lm    = lane & 15;
  const int bbase = blockIdx.x * 16;   // 16 batch rows per workgroup

  // ---- stage + convert weights (fp32 -> bf16), 128 threads ----
  for (int i = tid; i < G1 * H1; i += 128) w1h[i] = (__bf16)Whh1[i];
  for (int i = tid; i < G2 * H1; i += 128) w2x[i] = (__bf16)Wih2[i];
  for (int i = tid; i < G2 * H2; i += 128) w2h[i] = (__bf16)Whh2[i];
  for (int n = tid; n < G1; n += 128) {   // f32-WMMA B op: VGPR v holds W[n][2v+hi]
    const float* s = Wih1 + n * 4;
    w1xs[n * 4 + 0] = (__bf16)s[0];
    w1xs[n * 4 + 1] = (__bf16)s[2];
    w1xs[n * 4 + 2] = (__bf16)s[1];
    w1xs[n * 4 + 3] = (__bf16)s[3];
  }
  for (int i = tid; i < 2 * 16 * H1; i += 128) hb1[0][i] = (__bf16)0.0f;
  for (int i = tid; i < 2 * 16 * H2; i += 128) hb2[0][i] = (__bf16)0.0f;
  __syncthreads();

  // Wave w owns L1 hidden tile t4=w via gate quad {w, w+4, w+8, w+12};
  // waves 0..1 own L2 hidden tile t2=w via quad {w, w+2, w+4, w+6}.
  const int w2sel = w & 1;
  float b1q[4], b2q[4];
  int offw1[4], offw1x[4], offw2[4], offw2h[4];
#pragma unroll
  for (int q = 0; q < 4; ++q) {
    const int n1 = (w + 4 * q) * 16 + lm;
    b1q[q] = bih1[n1] + bhh1[n1];
    const int n2 = (w2sel + 2 * q) * 16 + lm;
    b2q[q] = bih2[n2] + bhh2[n2];
    offw1[q]  = n1 * H1 + hi * 16;     // halfword offsets, loop-invariant
    offw1x[q] = n1 * 4  + hi * 2;
    offw2[q]  = n2 * H1 + hi * 16;
    offw2h[q] = n2 * H2 + hi * 16;
  }

  v8f c1 = {};   // cell state, hidden tile w   (m = r+8*hi, n = lm+16*w)
  v8f c2 = {};   // layer-2 cell state (waves 0..1)

  const float4* xp = (const float4*)(x + (size_t)(bbase + lm) * T_SZ * 4);
  int p = 0;

#pragma unroll 1
  for (int t = 0; t < T_SZ; ++t) {
    asm volatile("" ::: "memory");     // keep LDS weight-tile loads inside loop

    const float4 xv = xp[t];
    __builtin_prefetch(xp + ((t + 8) & (T_SZ - 1)), 0, 1);
    v2f ax;
    ax.x = hi ? xv.z : xv.x;
    ax.y = hi ? xv.w : xv.y;

    // ---- phase 1: read h1_{t-1} ----
    const __bf16* h1cur = hb1[p] + lm * H1;
    v16bf ah0 = load_a_tile(h1cur, 16 * hi);
    v16bf ah1 = load_a_tile(h1cur, 32 + 16 * hi);

    // ---- phase 2: layer-1 gate quad (i,f,g,o) for hidden tile w ----
    v8f acc[4];
#pragma unroll
    for (int q = 0; q < 4; ++q) {
      v8f a;
#pragma unroll
      for (int r = 0; r < 8; ++r) a[r] = b1q[q];

      const __bf16* ps = &w1xs[offw1x[q]];
      v2f bx; bx.x = (float)ps[0]; bx.y = (float)ps[1];
      a = __builtin_amdgcn_wmma_f32_16x16x4_f32(
          false, ax, false, bx, (short)0, a, false, false);

      v16bf b0 = *(const v16bf*)&w1h[offw1[q]];
      a = __builtin_amdgcn_wmma_f32_16x16x32_bf16(
          false, ah0, false, b0, (short)0, a, false, false);

      v16bf b1 = *(const v16bf*)&w1h[offw1[q] + 32];
      a = __builtin_amdgcn_wmma_f32_16x16x32_bf16(
          false, ah1, false, b1, (short)0, a, false, false);

      acc[q] = a;
    }

    __bf16* h1n = hb1[p ^ 1];
#pragma unroll
    for (int r = 0; r < 8; ++r) {
      float ig = fast_sigmoid(acc[0][r]);
      float fg = fast_sigmoid(acc[1][r]);
      float gg = fast_tanh   (acc[2][r]);
      float og = fast_sigmoid(acc[3][r]);
      float c  = fg * c1[r] + ig * gg;
      c1[r] = c;
      h1n[(r + 8 * hi) * H1 + w * 16 + lm] = (__bf16)(og * fast_tanh(c));
    }

    // ---- phase 3: one barrier per step (ping-pong makes it sufficient) ----
    __syncthreads();

    // ---- phase 4: layer 2 on waves 0..1 ----
    if (w < 2) {
      const __bf16* h1new = hb1[p ^ 1] + lm * H1;
      v16bf a10 = load_a_tile(h1new, 16 * hi);
      v16bf a11 = load_a_tile(h1new, 32 + 16 * hi);
      v16bf a2  = load_a_tile(hb2[p] + lm * H2, 16 * hi);

      v8f acc2[4];
#pragma unroll
      for (int q = 0; q < 4; ++q) {
        v8f a;
#pragma unroll
        for (int r = 0; r < 8; ++r) a[r] = b2q[q];

        v16bf bx0 = *(const v16bf*)&w2x[offw2[q]];
        a = __builtin_amdgcn_wmma_f32_16x16x32_bf16(
            false, a10, false, bx0, (short)0, a, false, false);

        v16bf bx1 = *(const v16bf*)&w2x[offw2[q] + 32];
        a = __builtin_amdgcn_wmma_f32_16x16x32_bf16(
            false, a11, false, bx1, (short)0, a, false, false);

        v16bf bh = *(const v16bf*)&w2h[offw2h[q]];
        a = __builtin_amdgcn_wmma_f32_16x16x32_bf16(
            false, a2, false, bh, (short)0, a, false, false);

        acc2[q] = a;
      }

      __bf16* h2n = hb2[p ^ 1];
#pragma unroll
      for (int r = 0; r < 8; ++r) {
        float ig = fast_sigmoid(acc2[0][r]);
        float fg = fast_sigmoid(acc2[1][r]);
        float gg = fast_tanh   (acc2[2][r]);
        float og = fast_sigmoid(acc2[3][r]);
        float c  = fg * c2[r] + ig * gg;
        c2[r] = c;
        h2n[(r + 8 * hi) * H2 + w * 16 + lm] = (__bf16)(og * fast_tanh(c));
      }
    }

    p ^= 1;
  }

  __syncthreads();

  // ---- FC head on h2[:, T-1, :] : wave 0, lanes 0-15 own one batch row each ----
  if (w == 0) {
    float hrow[H2];
#pragma unroll
    for (int k = 0; k < H2; ++k) hrow[k] = (float)hb2[p][lm * H2 + k];

    float o = fc2b[0];
#pragma unroll
    for (int j = 0; j < FCN; ++j) {
      float s = fc1b[j];
#pragma unroll
      for (int k = 0; k < H2; ++k) s = fmaf(fc1w[j * H2 + k], hrow[k], s);
      s = fmaxf(s, 0.0f);
      o = fmaf(fc2w[j], s, o);
    }
    if (lane < 16) out[bbase + lm] = o;
  }
}

extern "C" void kernel_launch(void* const* d_in, const int* in_sizes, int n_in,
                              void* d_out, int out_size, void* d_ws, size_t ws_size,
                              hipStream_t stream) {
  (void)in_sizes; (void)n_in; (void)out_size; (void)d_ws; (void)ws_size;
  const float* x    = (const float*)d_in[0];
  const float* Wih1 = (const float*)d_in[1];
  const float* Whh1 = (const float*)d_in[2];
  const float* bih1 = (const float*)d_in[3];
  const float* bhh1 = (const float*)d_in[4];
  const float* Wih2 = (const float*)d_in[5];
  const float* Whh2 = (const float*)d_in[6];
  const float* bih2 = (const float*)d_in[7];
  const float* bhh2 = (const float*)d_in[8];
  const float* fc1w = (const float*)d_in[9];
  const float* fc1b = (const float*)d_in[10];
  const float* fc2w = (const float*)d_in[11];
  const float* fc2b = (const float*)d_in[12];

  lstm2_fused_wmma<<<dim3(B_SZ / 16), dim3(128), 0, stream>>>(
      x, Wih1, Whh1, bih1, bhh1, Wih2, Whh2, bih2, bhh2,
      fc1w, fc1b, fc2w, fc2b, (float*)d_out);
}
